// TransformerEncoderLayer_4990751998222
// MI455X (gfx1250) — matrix operand order repaired and checked by reading, not compiled, not run on hardware
//
#include <hip/hip_runtime.h>
#include <hip/hip_bf16.h>
#include <math.h>

// ---------------------------------------------------------------------------
// MI455X (gfx1250) implementation. Compute-bound on WMMA: all GEMMs and both
// attentions run on v_wmma_f32_16x16x32_f16 (f16 operands, fp32 accum).
// Working set (~100MB) fits in the 192MB L2; only the GCN edge scatter is
// memory/atomic bound. Requires ws_size >= ~102 MB.
//
// GEMM: 128x128 block tile, 8 waves, each wave 32x64 -> 8 WMMA per K-step
// against 12 ds_load_b128 (2 A-frags reused over 4 B-frags).
// ---------------------------------------------------------------------------

typedef __attribute__((ext_vector_type(16))) _Float16 v16h;
typedef __attribute__((ext_vector_type(8)))  _Float16 v8h;
typedef __attribute__((ext_vector_type(8)))  float    v8f;

#define DEVINL static __device__ __forceinline__

// ---- problem constants (from reference) ----
#define SEQ   2048
#define BATCH 2
#define DMOD  512
#define NHEAD 8
#define HID   512
#define XNH   4
#define PTOK  128
#define PDIM  4096
#define FFD   2048
#define NEDGE 65536
#define MROWS (SEQ * BATCH)   // 4096 flattened (s,b) rows

// ---------------------------------------------------------------------------
// WMMA helpers
// ---------------------------------------------------------------------------
DEVINL v8f do_wmma(v16h a, v16h b, v8f c) {
  return __builtin_amdgcn_wmma_f32_16x16x32_f16(false, a, false, b,
                                                (short)0, c, false, false);
}

// A-fragment (16x32 f16) from LDS. Layout: lane<16 -> M=lane, K = k0..k0+7 and
// k0+16..k0+23 with k0=0; lanes>=16 -> same M set, k0=8. Two ds_load_b128.
DEVINL v16h frag_a_lds(const _Float16* lds, int stride, int m, int kk, int lane) {
  int k0 = kk + ((lane & 16) ? 8 : 0);
  v8h lo = *(const v8h*)(lds + m * stride + k0);
  v8h hi = *(const v8h*)(lds + m * stride + k0 + 16);
  v16h r;
#pragma unroll
  for (int i = 0; i < 8; ++i) { r[i] = lo[i]; r[i + 8] = hi[i]; }
  return r;
}

// B-fragment (32x16 f16): lane<16 -> N=lane, K=0..7 and 16..23; lanes>=16 ->
// K=8..15 and 24..31. Source row-major [N][K], two 8-half loads per lane.
DEVINL v16h frag_b_lds(const _Float16* lds, int stride, int n, int kk, int lane) {
  int k0 = kk + ((lane & 16) ? 8 : 0);
  v8h lo = *(const v8h*)(lds + n * stride + k0);
  v8h hi = *(const v8h*)(lds + n * stride + k0 + 16);
  v16h r;
#pragma unroll
  for (int i = 0; i < 8; ++i) { r[i] = lo[i]; r[i + 8] = hi[i]; }
  return r;
}

// fp32x16 -> two v8h LDS stores (one 16-float strip per thread)
DEVINL void stage16(const float* __restrict__ gp, _Float16* lp) {
  const float4* p = (const float4*)gp;
  float4 a0 = p[0], a1 = p[1], a2 = p[2], a3 = p[3];
  v8h h0, h1;
  h0[0] = (_Float16)a0.x; h0[1] = (_Float16)a0.y;
  h0[2] = (_Float16)a0.z; h0[3] = (_Float16)a0.w;
  h0[4] = (_Float16)a1.x; h0[5] = (_Float16)a1.y;
  h0[6] = (_Float16)a1.z; h0[7] = (_Float16)a1.w;
  h1[0] = (_Float16)a2.x; h1[1] = (_Float16)a2.y;
  h1[2] = (_Float16)a2.z; h1[3] = (_Float16)a2.w;
  h1[4] = (_Float16)a3.x; h1[5] = (_Float16)a3.y;
  h1[6] = (_Float16)a3.z; h1[7] = (_Float16)a3.w;
  *(v8h*)lp       = h0;
  *(v8h*)(lp + 8) = h1;
}

// ---------------------------------------------------------------------------
// Generic GEMM: C[M,N] = A[M,K] @ W[N,K]^T (+bias)(+resid)(ReLU opt), fp32
// in/out, f16 WMMA compute. 256 threads = 8 waves; block tile 128x128; wave
// tile 32x64 (2 A-frags x 4 B-frags = 8 WMMA per 32-wide K-step).
// ---------------------------------------------------------------------------
template <bool RELU>
__global__ __launch_bounds__(256) void gemm_kernel(
    const float* __restrict__ A, int lda, const float* __restrict__ W, int ldw,
    const float* __restrict__ bias, const float* __restrict__ resid, int ldr,
    float* __restrict__ C, int ldc, int M, int N, int K) {
  __shared__ _Float16 As[128 * 32];   // 8 KB
  __shared__ _Float16 Ws[128 * 32];   // 8 KB

  const int tid  = threadIdx.x;
  const int lane = tid & 31;
  const int w    = tid >> 5;
  const int m0   = blockIdx.y * 128;
  const int n0   = blockIdx.x * 128;
  const int wm   = (w & 3) * 32;   // wave row base (2 x 16-row tiles)
  const int wn   = (w >> 2) * 64;  // wave col base (4 x 16-col tiles)

  const int lrow = tid >> 1;        // 0..127 (staging row)
  const int lk   = (tid & 1) * 16;  // 0,16   (staging k strip)

  v8f acc[2][4];
#pragma unroll
  for (int i = 0; i < 2; ++i)
#pragma unroll
    for (int j = 0; j < 4; ++j)
#pragma unroll
      for (int r = 0; r < 8; ++r) acc[i][j][r] = 0.0f;

  for (int kk = 0; kk < K; kk += 32) {
    // stage fp32 -> f16 tiles (A: 128x32, W: 128x32)
    stage16(A + (size_t)(m0 + lrow) * lda + kk + lk, As + lrow * 32 + lk);
    stage16(W + (size_t)(n0 + lrow) * ldw + kk + lk, Ws + lrow * 32 + lk);
    if (kk + 32 < K) {  // global_prefetch_b8 for next K tile
      __builtin_prefetch(A + (size_t)(m0 + lrow) * lda + kk + 32 + lk, 0, 1);
      __builtin_prefetch(W + (size_t)(n0 + lrow) * ldw + kk + 32 + lk, 0, 1);
    }
    __syncthreads();

    v16h af[2], bf[4];
#pragma unroll
    for (int i = 0; i < 2; ++i)
      af[i] = frag_a_lds(As, 32, wm + i * 16 + (lane & 15), 0, lane);
#pragma unroll
    for (int j = 0; j < 4; ++j)
      bf[j] = frag_b_lds(Ws, 32, wn + j * 16 + (lane & 15), 0, lane);
#pragma unroll
    for (int i = 0; i < 2; ++i)
#pragma unroll
      for (int j = 0; j < 4; ++j)
        acc[i][j] = do_wmma(af[i], bf[j], acc[i][j]);
    __syncthreads();
  }

  // epilogue: element (i,j,vgpr r,lane) -> row m0+wm+i*16+r+8*half,
  //                                        col n0+wn+j*16+(lane&15)
  const int rbase = m0 + wm + ((lane & 16) ? 8 : 0);
  const int cbase = n0 + wn + (lane & 15);
#pragma unroll
  for (int i = 0; i < 2; ++i) {
#pragma unroll
    for (int r = 0; r < 8; ++r) {
      int m = rbase + i * 16 + r;
#pragma unroll
      for (int j = 0; j < 4; ++j) {
        int n = cbase + j * 16;
        float v = acc[i][j][r];
        if (bias)  v += bias[n];
        if (resid) v += resid[(size_t)m * ldr + n];
        if (RELU)  v = fmaxf(v, 0.0f);
        C[(size_t)m * ldc + n] = v;
      }
    }
  }
}

// ---------------------------------------------------------------------------
// Flash attention (online softmax), templated on head dim. 128 threads = 4
// waves; block handles 64 query rows for one (batch, head); streams key tiles
// of 64. Scores and P@V both on WMMA; probs re-laid out C->A through LDS.
// ---------------------------------------------------------------------------
template <int HD>
__global__ __launch_bounds__(128) void flash_kernel(
    const float* __restrict__ Qg, int qB, int qSeq, int qH,
    const float* __restrict__ Kg, int kB, int kSeq, int kH,
    const float* __restrict__ Vg, int vB, int vSeq, int vH,
    float* __restrict__ Og, int oB, int oSeq, int oH,
    int T, float scale, const unsigned char* __restrict__ maskp, int maskB,
    int nh) {
  constexpr int KT = 64;
  constexpr int NK = HD / 32;  // score K-steps
  constexpr int NO = HD / 16;  // output col tiles
  __shared__ _Float16 ks[KT * HD];       // keys, row-major [t][d]
  __shared__ _Float16 vt[HD * KT];       // V transposed  [d][t]
  __shared__ _Float16 pr[4 * 16 * KT];   // per-wave probs [16][KT]

  const int tid  = threadIdx.x;
  const int lane = tid & 31;
  const int w    = tid >> 5;
  const int b    = blockIdx.y / nh;
  const int h    = blockIdx.y % nh;
  const int s0   = blockIdx.x * 64;

  const size_t qoff = (size_t)b * qB + (size_t)h * qH;
  const size_t koff = (size_t)b * kB + (size_t)h * kH;
  const size_t voff = (size_t)b * vB + (size_t)h * vH;
  const size_t ooff = (size_t)b * oB + (size_t)h * oH;

  // Q fragments, loaded once (A layout straight from global fp32)
  v16h qf[NK];
  {
    const int m  = w * 16 + (lane & 15);
    const int k0 = (lane & 16) ? 8 : 0;
    const float* qp = Qg + qoff + (size_t)(s0 + m) * qSeq;
#pragma unroll
    for (int d = 0; d < NK; ++d) {
      const float4* p0 = (const float4*)(qp + d * 32 + k0);
      const float4* p1 = (const float4*)(qp + d * 32 + k0 + 16);
      float4 x0 = p0[0], x1 = p0[1], y0 = p1[0], y1 = p1[1];
      v16h f;
      f[0]  = (_Float16)x0.x; f[1]  = (_Float16)x0.y;
      f[2]  = (_Float16)x0.z; f[3]  = (_Float16)x0.w;
      f[4]  = (_Float16)x1.x; f[5]  = (_Float16)x1.y;
      f[6]  = (_Float16)x1.z; f[7]  = (_Float16)x1.w;
      f[8]  = (_Float16)y0.x; f[9]  = (_Float16)y0.y;
      f[10] = (_Float16)y0.z; f[11] = (_Float16)y0.w;
      f[12] = (_Float16)y1.x; f[13] = (_Float16)y1.y;
      f[14] = (_Float16)y1.z; f[15] = (_Float16)y1.w;
      qf[d] = f;
    }
  }

  float rmax[8], rsum[8];
#pragma unroll
  for (int r = 0; r < 8; ++r) { rmax[r] = -3.0e38f; rsum[r] = 0.0f; }
  v8f oacc[NO];
#pragma unroll
  for (int j = 0; j < NO; ++j)
#pragma unroll
    for (int r = 0; r < 8; ++r) oacc[j][r] = 0.0f;

  for (int t0 = 0; t0 < T; t0 += KT) {
    __syncthreads();  // previous tile fully consumed
    for (int idx = tid; idx < KT * HD; idx += 128) {
      int t = idx / HD, d = idx % HD;
      ks[t * HD + d] = (_Float16)Kg[koff + (size_t)(t0 + t) * kSeq + d];
      vt[d * KT + t] = (_Float16)Vg[voff + (size_t)(t0 + t) * vSeq + d];
    }
    __syncthreads();

    // scores: four 16x16 tiles per wave
    float sc[4][8];
#pragma unroll
    for (int ct = 0; ct < 4; ++ct) {
      v8f a;
#pragma unroll
      for (int r = 0; r < 8; ++r) a[r] = 0.0f;
#pragma unroll
      for (int d = 0; d < NK; ++d) {
        v16h bf = frag_b_lds(ks, HD, ct * 16 + (lane & 15), d * 32, lane);
        a = do_wmma(qf[d], bf, a);
      }
      bool mok = true;
      if (maskp) {
        int t = t0 + ct * 16 + (lane & 15);
        mok = maskp[(size_t)b * maskB + t] != 0;
      }
#pragma unroll
      for (int r = 0; r < 8; ++r) sc[ct][r] = mok ? a[r] * scale : -1.0e9f;
    }

    // online softmax: per-row reductions over 16-lane halves
    float nmax[8], corr[8];
#pragma unroll
    for (int r = 0; r < 8; ++r) {
      float v = fmaxf(fmaxf(sc[0][r], sc[1][r]), fmaxf(sc[2][r], sc[3][r]));
#pragma unroll
      for (int off = 8; off >= 1; off >>= 1)
        v = fmaxf(v, __shfl_xor(v, off, 32));
      nmax[r] = fmaxf(rmax[r], v);
      corr[r] = __expf(rmax[r] - nmax[r]);
      rmax[r] = nmax[r];
    }
#pragma unroll
    for (int r = 0; r < 8; ++r) {
      float s = 0.0f;
#pragma unroll
      for (int ct = 0; ct < 4; ++ct) {
        float p = __expf(sc[ct][r] - nmax[r]);
        sc[ct][r] = p;
        s += p;
      }
#pragma unroll
      for (int off = 8; off >= 1; off >>= 1) s += __shfl_xor(s, off, 32);
      rsum[r] = rsum[r] * corr[r] + s;
#pragma unroll
      for (int j = 0; j < NO; ++j) oacc[j][r] *= corr[r];
    }

    // write probs (C layout) -> per-wave LDS, reread as A fragments
    {
      _Float16* pw = pr + w * (16 * KT);
      const int mr = (lane & 16) ? 8 : 0;
      const int cc = lane & 15;
#pragma unroll
      for (int ct = 0; ct < 4; ++ct)
#pragma unroll
        for (int r = 0; r < 8; ++r)
          pw[(mr + r) * KT + ct * 16 + cc] = (_Float16)sc[ct][r];
    }
    __syncthreads();

    // P @ V
    {
      const _Float16* pw = pr + w * (16 * KT);
#pragma unroll
      for (int kt = 0; kt < KT / 32; ++kt) {
        v16h pa = frag_a_lds(pw, KT, lane & 15, kt * 32, lane);
#pragma unroll
        for (int j = 0; j < NO; ++j) {
          v16h vb = frag_b_lds(vt, KT, j * 16 + (lane & 15), kt * 32, lane);
          oacc[j] = do_wmma(pa, vb, oacc[j]);
        }
      }
    }
  }

  // epilogue
  const int srow = s0 + w * 16 + ((lane & 16) ? 8 : 0);
  const int dcol = lane & 15;
#pragma unroll
  for (int r = 0; r < 8; ++r) {
    float inv = 1.0f / rsum[r];
#pragma unroll
    for (int j = 0; j < NO; ++j)
      Og[ooff + (size_t)(srow + r) * oSeq + j * 16 + dcol] = oacc[j][r] * inv;
  }
}

// ---------------------------------------------------------------------------
// LayerNorm over D=512: one 256-thread block per row, 2 elems/thread.
// ---------------------------------------------------------------------------
__global__ __launch_bounds__(256) void ln_kernel(
    const float* __restrict__ x, const float* __restrict__ g,
    const float* __restrict__ bt, float* __restrict__ out) {
  __shared__ float red[256];
  const int row = blockIdx.x;
  const int tid = threadIdx.x;
  const float* xr = x + (size_t)row * 512;
  float v0 = xr[tid], v1 = xr[tid + 256];
  red[tid] = v0 + v1;
  __syncthreads();
  for (int o = 128; o > 0; o >>= 1) {
    if (tid < o) red[tid] += red[tid + o];
    __syncthreads();
  }
  float mu = red[0] * (1.0f / 512.0f);
  __syncthreads();
  float d0 = v0 - mu, d1 = v1 - mu;
  red[tid] = d0 * d0 + d1 * d1;
  __syncthreads();
  for (int o = 128; o > 0; o >>= 1) {
    if (tid < o) red[tid] += red[tid + o];
    __syncthreads();
  }
  float rstd = rsqrtf(red[0] * (1.0f / 512.0f) + 1e-5f);
  out[(size_t)row * 512 + tid]       = d0 * rstd * g[tid] + bt[tid];
  out[(size_t)row * 512 + tid + 256] = d1 * rstd * g[tid + 256] + bt[tid + 256];
}

// ---------------------------------------------------------------------------
// Small helper kernels
// ---------------------------------------------------------------------------
__global__ void zero_kernel(float* p, int n) {
  int i = blockIdx.x * 256 + threadIdx.x;
  if (i < n) p[i] = 0.0f;
}

__global__ void deg_kernel(const int* __restrict__ dst, float* __restrict__ deg,
                           int E) {
  int i = blockIdx.x * 256 + threadIdx.x;
  if (i < E) atomicAdd(&deg[dst[i]], 1.0f);
}

__global__ void dinv_kernel(const float* __restrict__ deg,
                            float* __restrict__ dinv, int n) {
  int i = blockIdx.x * 256 + threadIdx.x;
  if (i < n) dinv[i] = rsqrtf(deg[i] + 1.0f);  // +1 self loop; deg>=1 always
}

// Scatter-add GCN messages (edges only; self-loops fused in gcn_post).
__global__ void scatter_kernel(const int* __restrict__ srci,
                               const int* __restrict__ dsti,
                               const float* __restrict__ dinv,
                               const float* __restrict__ hw,
                               float* __restrict__ agg, int E) {
  const int nch = HID / 4;  // 128 float4 chunks
  int i = blockIdx.x * 256 + threadIdx.x;
  if (i >= E * nch) return;
  int e = i / nch;
  int c = (i % nch) * 4;
  int s = srci[e], t = dsti[e];
  float wgt = dinv[s] * dinv[t];
#pragma unroll
  for (int bb = 0; bb < BATCH; ++bb) {
    const float4 hv = *(const float4*)(hw + ((size_t)s * BATCH + bb) * HID + c);
    float* dp = agg + ((size_t)t * BATCH + bb) * HID + c;
    atomicAdd(dp + 0, hv.x * wgt);
    atomicAdd(dp + 1, hv.y * wgt);
    atomicAdd(dp + 2, hv.z * wgt);
    atomicAdd(dp + 3, hv.w * wgt);
  }
}

// h = relu(agg + hw * dinv[s]^2 + gcn_b)   (self-loop + bias + relu)
__global__ void gcn_post_kernel(const float* __restrict__ agg,
                                const float* __restrict__ hw,
                                const float* __restrict__ dinv,
                                const float* __restrict__ gb,
                                float* __restrict__ h, int total) {
  int i = blockIdx.x * 256 + threadIdx.x;
  if (i >= total) return;
  int c = i & (HID - 1);
  int s = i >> 10;  // / (HID*BATCH) = /1024
  float di = dinv[s];
  float v = agg[i] + hw[i] * di * di + gb[c];
  h[i] = fmaxf(v, 0.0f);
}

__global__ void add_kernel(const float* __restrict__ a,
                           const float* __restrict__ b, float* __restrict__ o,
                           int n) {
  int i = blockIdx.x * 256 + threadIdx.x;
  if (i < n) o[i] = a[i] + b[i];
}

// ---------------------------------------------------------------------------
// Orchestration
// ---------------------------------------------------------------------------
extern "C" void kernel_launch(void* const* d_in, const int* in_sizes, int n_in,
                              void* d_out, int out_size, void* d_ws,
                              size_t ws_size, hipStream_t stream) {
  const float* src    = (const float*)d_in[0];   // [S,B,D] -> [4096,512]
  const float* prompt = (const float*)d_in[1];   // [B,P,PD] -> [256,4096]
  const float* aiw = (const float*)d_in[2];  const float* aib = (const float*)d_in[3];
  const float* aow = (const float*)d_in[4];  const float* aob = (const float*)d_in[5];
  const float* liw = (const float*)d_in[6];  const float* lib = (const float*)d_in[7];
  const float* gcw = (const float*)d_in[8];  const float* gcb = (const float*)d_in[9];
  const float* xqw = (const float*)d_in[10]; const float* xqb = (const float*)d_in[11];
  const float* xkw = (const float*)d_in[12]; const float* xkb = (const float*)d_in[13];
  const float* xvw = (const float*)d_in[14]; const float* xvb = (const float*)d_in[15];
  const float* xow = (const float*)d_in[16]; const float* xob = (const float*)d_in[17];
  const float* low = (const float*)d_in[18]; const float* lob = (const float*)d_in[19];
  const float* l1w = (const float*)d_in[20]; const float* l1b = (const float*)d_in[21];
  const float* l2w = (const float*)d_in[22]; const float* l2b = (const float*)d_in[23];
  const float* n1lg = (const float*)d_in[24]; const float* n1lb = (const float*)d_in[25];
  const float* n1gg = (const float*)d_in[26]; const float* n1gb = (const float*)d_in[27];
  const float* n2g  = (const float*)d_in[28]; const float* n2b  = (const float*)d_in[29];
  const int* eidx = (const int*)d_in[30];                // [2,E]
  const unsigned char* pmask = (const unsigned char*)d_in[31];  // [B,P] bool

  float* ws = (float*)d_ws;
  const size_t M2 = 2097152ULL;               // 2M floats (4096x512)
  float* deg  = ws;                           // 2048
  float* dinv = ws + 2048;                    // 2048
  float* BIG  = ws + 4096;                    // 8M floats: qkv, later ff1
  float* R0 = BIG + 8388608;                  // o_self, later hcomb
  float* R1 = R0 + M2;                        // h_pfn (persistent)
  float* R2 = R1 + M2;                        // hlin, later lin_out pre-LN
  float* R3 = R2 + M2;                        // hw,   later h_local (LN out)
  float* R4 = R3 + M2;                        // agg / h (persistent)
  float* R5 = R4 + M2;                        // q,    later FFN pre-LN
  float* R6 = R5 + M2;                        // cross-attn out
  float* R7 = R6 + M2;                        // attn proj pre-LN, later h2x
  float* KB = R7 + M2;                        // k buf [256,512]
  float* VB = KB + 131072;                    // v buf [256,512]
  (void)in_sizes; (void)n_in; (void)out_size; (void)ws_size;

  dim3 blk256(256), blk128(128);

  // ---- GCN degree normalization ----
  zero_kernel<<<8, blk256, 0, stream>>>(deg, SEQ);
  deg_kernel<<<NEDGE / 256, blk256, 0, stream>>>(eidx + NEDGE, deg, NEDGE);
  dinv_kernel<<<8, blk256, 0, stream>>>(deg, dinv, SEQ);

  // ---- global self-attention branch ----
  gemm_kernel<false><<<dim3(12, 32), blk256, 0, stream>>>(
      src, DMOD, aiw, DMOD, aib, nullptr, 0, BIG, 3 * DMOD, MROWS, 3 * DMOD, DMOD);
  flash_kernel<64><<<dim3(32, BATCH * NHEAD), blk128, 0, stream>>>(
      BIG, 3 * DMOD, BATCH * 3 * DMOD, 64,              // Q
      BIG + DMOD, 3 * DMOD, BATCH * 3 * DMOD, 64,       // K
      BIG + 2 * DMOD, 3 * DMOD, BATCH * 3 * DMOD, 64,   // V
      R0, DMOD, BATCH * DMOD, 64,                        // O
      SEQ, 0.125f, nullptr, 0, NHEAD);
  gemm_kernel<false><<<dim3(4, 32), blk256, 0, stream>>>(
      R0, DMOD, aow, DMOD, aob, src, DMOD, R7, DMOD, MROWS, DMOD, DMOD);
  ln_kernel<<<MROWS, blk256, 0, stream>>>(R7, n1gg, n1gb, R1);  // h_pfn

  // ---- local GNN branch ----
  gemm_kernel<false><<<dim3(4, 32), blk256, 0, stream>>>(
      src, DMOD, liw, DMOD, lib, nullptr, 0, R2, HID, MROWS, HID, DMOD);
  gemm_kernel<false><<<dim3(4, 32), blk256, 0, stream>>>(
      R2, HID, gcw, HID, nullptr, nullptr, 0, R3, HID, MROWS, HID, HID);
  zero_kernel<<<(int)(M2 / 256), blk256, 0, stream>>>(R4, (int)M2);
  scatter_kernel<<<NEDGE * (HID / 4) / 256, blk256, 0, stream>>>(
      eidx, eidx + NEDGE, dinv, R3, R4, NEDGE);
  gcn_post_kernel<<<(int)(M2 / 256), blk256, 0, stream>>>(
      R4, R3, dinv, gcb, R4, (int)M2);

  // cross-attention: nodes attend to prompt
  gemm_kernel<false><<<dim3(4, 32), blk256, 0, stream>>>(
      R4, HID, xqw, HID, xqb, nullptr, 0, R5, HID, MROWS, HID, HID);
  gemm_kernel<false><<<dim3(4, 2), blk256, 0, stream>>>(
      prompt, PDIM, xkw, PDIM, xkb, nullptr, 0, KB, HID, BATCH * PTOK, HID, PDIM);
  gemm_kernel<false><<<dim3(4, 2), blk256, 0, stream>>>(
      prompt, PDIM, xvw, PDIM, xvb, nullptr, 0, VB, HID, BATCH * PTOK, HID, PDIM);
  flash_kernel<128><<<dim3(32, BATCH * XNH), blk128, 0, stream>>>(
      R5, HID, BATCH * HID, 128,          // Q: rows (s*B+b)
      KB, PTOK * HID, HID, 128,           // K: rows (b*P+t)
      VB, PTOK * HID, HID, 128,           // V
      R6, HID, BATCH * HID, 128,          // O
      PTOK, 0.08838834764831845f, pmask, PTOK, XNH);
  gemm_kernel<false><<<dim3(4, 32), blk256, 0, stream>>>(
      R6, HID, xow, HID, xob, R4, HID, R7, HID, MROWS, HID, HID);  // h2x = h + xo(attn)
  gemm_kernel<false><<<dim3(4, 32), blk256, 0, stream>>>(
      R7, HID, low, HID, lob, src, DMOD, R2, DMOD, MROWS, DMOD, HID);
  ln_kernel<<<MROWS, blk256, 0, stream>>>(R2, n1lg, n1lb, R3);  // h_local

  // ---- combine + FFN ----
  add_kernel<<<(int)(M2 / 256), blk256, 0, stream>>>(R3, R1, R0, (int)M2);
  gemm_kernel<true><<<dim3(16, 32), blk256, 0, stream>>>(
      R0, DMOD, l1w, DMOD, l1b, nullptr, 0, BIG, FFD, MROWS, FFD, DMOD);
  gemm_kernel<false><<<dim3(4, 32), blk256, 0, stream>>>(
      BIG, FFD, l2w, FFD, l2b, R0, DMOD, R5, DMOD, MROWS, DMOD, FFD);
  ln_kernel<<<MROWS, blk256, 0, stream>>>(R5, n2g, n2b, (float*)d_out);
}
